// PAB3D_51196010168769
// MI455X (gfx1250) — matrix operand order repaired and checked by reading, not compiled, not run on hardware
//
#include <hip/hip_runtime.h>

typedef __bf16 bf16;
typedef __attribute__((ext_vector_type(16))) __bf16 v16bf;
typedef __attribute__((ext_vector_type(8)))  float  v8f;

// ---------------- fragment helpers (wave32 WMMA 16x16x32 bf16) ----------------
// A-matrix (16x32, 16-bit): lane L: row M=L&15, khalf=L>>4.
//   element e=2v+j -> k = 16*(v>>2) + 8*khalf + (e&7)  => p[0..7], p[16..23] at base+khalf*8
// B-matrix (32x16, 16-bit): lane L: col N=L&15, khalf=L>>4.
//   element e -> k = 16*(e>>3) + 8*khalf + (e&7)  => p[0..7], p[16..23] at base+khalf*8
// C/D (16x16 f32): lane L: col=L&15, row = v + 8*(L>>4)

__device__ __forceinline__ v16bf frag_a(const bf16* p) {  // p = row_base + khalf*8
    union { v16bf v; uint4 u[2]; } f;
    f.u[0] = *reinterpret_cast<const uint4*>(p);
    f.u[1] = *reinterpret_cast<const uint4*>(p + 16);
    return f.v;
}
__device__ __forceinline__ v16bf frag_b(const bf16* p) {  // p = col_base + khalf*16; loads base+khalf*8 and +16
    union { v16bf v; uint4 u[2]; } f;
    const bf16* q = p - 8 * ((threadIdx.x >> 4) & 1);
    f.u[0] = *reinterpret_cast<const uint4*>(q);
    f.u[1] = *reinterpret_cast<const uint4*>(q + 16);
    return f.v;
}
__device__ __forceinline__ v16bf frag_zero() {
    union { v16bf v; uint4 u[2]; } f;
    f.u[0] = make_uint4(0u, 0u, 0u, 0u);
    f.u[1] = make_uint4(0u, 0u, 0u, 0u);
    return f.v;
}
__device__ __forceinline__ v8f vzero8() {
    v8f z;
#pragma unroll
    for (int i = 0; i < 8; ++i) z[i] = 0.0f;
    return z;
}
__device__ __forceinline__ v8f wmma_bf16(v16bf a, v16bf b, v8f c) {
    return __builtin_amdgcn_wmma_f32_16x16x32_bf16(false, a, false, b, (short)0, c, false, false);
}

// ordered-uint encoding for float atomicMax
__device__ __forceinline__ unsigned f2o(float f) {
    unsigned u = __float_as_uint(f);
    return (u & 0x80000000u) ? ~u : (u | 0x80000000u);
}
__device__ __forceinline__ float o2f(unsigned u) {
    unsigned b = (u & 0x80000000u) ? (u & 0x7fffffffu) : ~u;
    return __uint_as_float(b);
}

// ---------------- constants ----------------
// B=2, C=P=64, N=4096 (16x16x16)

// ---------------- init scalars ----------------
__global__ void k_init(unsigned* Mu, float* Zf) {
    int t = threadIdx.x;
    if (t < 2) { Mu[t] = 0u; Zf[t] = 0.0f; }
}

// ---------------- weight prep: [O][I][27] f32 -> [27][O][I] bf16 ----------------
__global__ void k_wprep(const float* __restrict__ wb, const float* __restrict__ wo,
                        bf16* __restrict__ wkb, bf16* __restrict__ wko) {
    int g = blockIdx.x * blockDim.x + threadIdx.x;
    if (g >= 27 * 64 * 64) return;
    int off = g >> 12;          // g / 4096
    int co  = (g >> 6) & 63;
    int ci  = g & 63;
    int src = (co * 64 + ci) * 27 + off;
    wkb[g] = (bf16)wb[src];
    wko[g] = (bf16)wo[src];
}

// ---------------- projections: 1x1x1 convs + bf16 transpose of x ----------------
// topT/centerT: [B][N][64] bf16 ([col][k] layout for both GEMM1 operands)
// xT: [B][N][64] bf16 (conv implicit-GEMM B operand)
__global__ void __launch_bounds__(256) k_proj(
        const float* __restrict__ x,
        const float* __restrict__ tw, const float* __restrict__ tb,
        const float* __restrict__ cw, const float* __restrict__ cb,
        bf16* __restrict__ topT, bf16* __restrict__ centerT, bf16* __restrict__ xT) {
    int g = blockIdx.x * blockDim.x + threadIdx.x;   // b*4096 + n
    int b = g >> 12;
    int n = g & 4095;
    const float* xp = x + (size_t)b * 262144 + n;
    float xv[64];
#pragma unroll
    for (int c = 0; c < 64; ++c) xv[c] = xp[(size_t)c * 4096];
    bf16* xo = xT + (size_t)g * 64;
#pragma unroll
    for (int c = 0; c < 64; ++c) xo[c] = (bf16)xv[c];
    bf16* to = topT + (size_t)g * 64;
    bf16* ce = centerT + (size_t)g * 64;
    for (int p = 0; p < 64; ++p) {
        float s1 = tb[p], s2 = cb[p];
#pragma unroll
        for (int c = 0; c < 64; ++c) {
            s1 = fmaf(tw[p * 64 + c], xv[c], s1);
            s2 = fmaf(cw[p * 64 + c], xv[c], s2);
        }
        to[p] = (bf16)s1;
        ce[p] = (bf16)s2;
    }
}

// ---------------- 3x3x3 conv as 27-offset implicit GEMM (WMMA) ----------------
// in : [B][N][64] bf16 (spatial-major, channel contiguous)
// wk : [27][64][64] bf16 ([off][co][ci])
// out: [B][64][N]  (bf16 or f32)
template <bool BF16OUT>
__global__ void __launch_bounds__(128) k_conv3(
        const bf16* __restrict__ xin, const bf16* __restrict__ wk,
        const float* __restrict__ bias,
        bf16* __restrict__ obf, float* __restrict__ of) {
    int blk = blockIdx.x;            // b*256 + ntile
    int b   = blk >> 8;
    int n0  = (blk & 255) << 4;
    int lane = threadIdx.x & 31;
    int wv   = threadIdx.x >> 5;     // co-tile 0..3
    int l15  = lane & 15;
    int kh   = lane >> 4;

    const bf16* xb = xin + (size_t)b * 4096 * 64;
    int s = n0 + l15;                // this lane's output column (B operand)
    int h = (s >> 8) & 15, w = (s >> 4) & 15, d = s & 15;

    v8f acc = vzero8();
    for (int off = 0; off < 27; ++off) {
        int dz = off / 9 - 1;
        int dy = (off / 3) % 3 - 1;
        int dx = off % 3 - 1;
        int hh = h + dz, ww = w + dy, dd = d + dx;
        bool ok = ((unsigned)hh < 16u) && ((unsigned)ww < 16u) && ((unsigned)dd < 16u);
        int sp = (hh << 8) | (ww << 4) | dd;
        const bf16* xrow = xb + (size_t)sp * 64;
        const bf16* wrow = wk + ((size_t)off * 64 + wv * 16 + l15) * 64;  // A row for this lane
#pragma unroll
        for (int ch = 0; ch < 2; ++ch) {          // ci halves: 0..31, 32..63
            v16bf bfr = ok ? frag_b(xrow + ch * 32 + kh * 16) : frag_zero();
            v16bf afr = frag_a(wrow + ch * 32 + kh * 8);
            acc = wmma_bf16(afr, bfr, acc);
        }
    }
#pragma unroll
    for (int v = 0; v < 8; ++v) {
        int co = wv * 16 + v + 8 * kh;
        float val = acc[v] + bias[co];
        size_t oidx = ((size_t)b * 64 + co) * 4096 + s;
        if (BF16OUT) obf[oidx] = (bf16)val; else of[oidx] = val;
    }
}

// ---------------- pass 1: sp = centerT x top, global max per batch ----------------
__global__ void __launch_bounds__(256) k_max(
        const bf16* __restrict__ centerT, const bf16* __restrict__ topT,
        unsigned* __restrict__ Mu) {
    int wv = threadIdx.x >> 5;
    int gw = blockIdx.x * 8 + wv;    // 0..511
    int b  = gw >> 8;
    int n0 = (gw & 255) << 4;
    int lane = threadIdx.x & 31;
    int l15 = lane & 15, kh = lane >> 4;

    const bf16* cT = centerT + (size_t)b * 4096 * 64;
    const bf16* tT = topT    + (size_t)b * 4096 * 64;
    v16bf a0 = frag_a(cT + (size_t)(n0 + l15) * 64 +  0 + kh * 8);
    v16bf a1 = frag_a(cT + (size_t)(n0 + l15) * 64 + 32 + kh * 8);

    float mx = -3.0e38f;
    for (int m0 = 0; m0 < 4096; m0 += 16) {
        const bf16* trow = tT + (size_t)(m0 + l15) * 64;
        v8f dacc = vzero8();
        dacc = wmma_bf16(a0, frag_b(trow +  0 + kh * 16), dacc);
        dacc = wmma_bf16(a1, frag_b(trow + 32 + kh * 16), dacc);
#pragma unroll
        for (int v = 0; v < 8; ++v) mx = fmaxf(mx, dacc[v]);
    }
    for (int o = 16; o; o >>= 1) mx = fmaxf(mx, __shfl_xor(mx, o));
    if (lane == 0) atomicMax(Mu + b, f2o(mx));
}

// ---------------- pass 2: recompute sp, exp, Z, fused GEMM2 ----------------
// attn[b][n][c] = sum_m exp(sp[n,m]-M) * bottom[c][m]   (unnormalized)
// Each wave owns a private 16x64 E-tile in LDS; same-wave LDS ops are in-order
// at the LDS unit (DScnt), so only wavefront-scope fences (no block barriers)
// are needed for the D-fragment -> A-fragment transpose round-trip.
__global__ void __launch_bounds__(256) k_av(
        const bf16* __restrict__ centerT, const bf16* __restrict__ topT,
        const bf16* __restrict__ bot,     // [B][64][N] bf16
        const unsigned* __restrict__ Mu, float* __restrict__ Zf,
        float* __restrict__ attn) {
    __shared__ __align__(16) bf16 elds[8][16][64];   // 16 KB, per-wave private tiles

    int wv = threadIdx.x >> 5;
    int gw = blockIdx.x * 8 + wv;
    int b  = gw >> 8;
    int n0 = (gw & 255) << 4;
    int lane = threadIdx.x & 31;
    int l15 = lane & 15, kh = lane >> 4;

    const bf16* cT = centerT + (size_t)b * 4096 * 64;
    const bf16* tT = topT    + (size_t)b * 4096 * 64;
    const bf16* bb = bot     + (size_t)b * 64 * 4096;
    float Mg = o2f(Mu[b]);

    v16bf a0 = frag_a(cT + (size_t)(n0 + l15) * 64 +  0 + kh * 8);
    v16bf a1 = frag_a(cT + (size_t)(n0 + l15) * 64 + 32 + kh * 8);

    float zs = 0.0f;
    v8f oacc[4];
#pragma unroll
    for (int ct = 0; ct < 4; ++ct) oacc[ct] = vzero8();

    for (int m0 = 0; m0 < 4096; m0 += 64) {
        // GEMM1: four 16x16 sp tiles covering columns m0..m0+63
        v8f d[4];
#pragma unroll
        for (int t = 0; t < 4; ++t) {
            const bf16* trow = tT + (size_t)(m0 + t * 16 + l15) * 64;
            v8f acc = vzero8();
            acc = wmma_bf16(a0, frag_b(trow +  0 + kh * 16), acc);
            acc = wmma_bf16(a1, frag_b(trow + 32 + kh * 16), acc);
            d[t] = acc;
        }
        // exp + Z partial + write E tile (bf16) to this wave's LDS region
#pragma unroll
        for (int t = 0; t < 4; ++t) {
#pragma unroll
            for (int v = 0; v < 8; ++v) {
                float e = __expf(d[t][v] - Mg);
                zs += e;
                elds[wv][v + 8 * kh][t * 16 + l15] = (bf16)e;
            }
        }
        __builtin_amdgcn_fence(__ATOMIC_ACQ_REL, "wavefront");
        // reload E tile as two A-fragments (K = m0..m0+31, m0+32..m0+63)
        v16bf ae0 = frag_a(&elds[wv][l15][0]  + kh * 8);
        v16bf ae1 = frag_a(&elds[wv][l15][32] + kh * 8);
        __builtin_amdgcn_fence(__ATOMIC_ACQ_REL, "wavefront");
        // GEMM2: accumulate E x bottom^T over this 64-wide m chunk
#pragma unroll
        for (int ct = 0; ct < 4; ++ct) {
            const bf16* brow = bb + (size_t)(ct * 16 + l15) * 4096 + m0;
            oacc[ct] = wmma_bf16(ae0, frag_b(brow +  0 + kh * 16), oacc[ct]);
            oacc[ct] = wmma_bf16(ae1, frag_b(brow + 32 + kh * 16), oacc[ct]);
        }
    }
#pragma unroll
    for (int ct = 0; ct < 4; ++ct)
#pragma unroll
        for (int v = 0; v < 8; ++v)
            attn[((size_t)b * 4096 + n0 + v + 8 * kh) * 64 + ct * 16 + l15] = oacc[ct][v];

    for (int o = 16; o; o >>= 1) zs += __shfl_xor(zs, o);
    if (lane == 0) atomicAdd(Zf + b, zs);
}

// ---------------- z = x + attn/Z (flat reinterpret == reshape), transposed bf16 ----------------
__global__ void k_zadd(const float* __restrict__ x, const float* __restrict__ attn,
                       const float* __restrict__ Zf, bf16* __restrict__ zT) {
    int g = blockIdx.x * blockDim.x + threadIdx.x;   // (b*4096+s)*64 + c
    int c = g & 63;
    int s = (g >> 6) & 4095;
    int b = g >> 18;
    float invZ = 1.0f / Zf[b];
    size_t lin = (size_t)b * 262144 + (size_t)c * 4096 + s;   // [B,N,C] flat == [B,C,HWD] flat
    zT[g] = (bf16)(x[lin] + attn[lin] * invZ);
}

// ---------------- launcher ----------------
extern "C" void kernel_launch(void* const* d_in, const int* in_sizes, int n_in,
                              void* d_out, int out_size, void* d_ws, size_t ws_size,
                              hipStream_t stream) {
    (void)in_sizes; (void)n_in; (void)out_size; (void)ws_size;
    const float* x   = (const float*)d_in[0];
    const float* twt = (const float*)d_in[1];
    const float* tbb = (const float*)d_in[2];
    const float* cwt = (const float*)d_in[3];
    const float* cbb = (const float*)d_in[4];
    const float* bwt = (const float*)d_in[5];
    const float* bbb = (const float*)d_in[6];
    const float* owt = (const float*)d_in[7];
    const float* obb = (const float*)d_in[8];
    float* out = (float*)d_out;

    char* ws = (char*)d_ws;
    size_t off = 0;
    auto carve = [&](size_t bytes) -> char* {
        char* p = ws + off;
        off += (bytes + 255) & ~(size_t)255;
        return p;
    };
    const size_t BNC = (size_t)2 * 4096 * 64;
    bf16*     topT    = (bf16*)carve(BNC * 2);          // [B][N][64]
    bf16*     centerT = (bf16*)carve(BNC * 2);          // [B][N][64]
    bf16*     xT      = (bf16*)carve(BNC * 2);          // [B][N][64]
    bf16*     bot     = (bf16*)carve(BNC * 2);          // [B][64][N]
    bf16*     zT      = (bf16*)carve(BNC * 2);          // [B][N][64]
    float*    attn    = (float*)carve(BNC * 4);         // [B][N][64] unnormalized
    bf16*     wkb     = (bf16*)carve((size_t)27 * 4096 * 2);
    bf16*     wko     = (bf16*)carve((size_t)27 * 4096 * 2);
    unsigned* Mu      = (unsigned*)carve(64);
    float*    Zf      = (float*)carve(64);

    k_init<<<1, 32, 0, stream>>>(Mu, Zf);
    k_wprep<<<(27 * 4096 + 255) / 256, 256, 0, stream>>>(bwt, owt, wkb, wko);
    k_proj<<<32, 256, 0, stream>>>(x, twt, tbb, cwt, cbb, topT, centerT, xT);
    k_conv3<true><<<512, 128, 0, stream>>>(xT, wkb, bbb, bot, nullptr);
    k_max<<<64, 256, 0, stream>>>(centerT, topT, Mu);
    k_av<<<64, 256, 0, stream>>>(centerT, topT, bot, Mu, Zf, attn);
    k_zadd<<<2048, 256, 0, stream>>>(x, attn, Zf, zT);
    k_conv3<false><<<512, 128, 0, stream>>>(zT, wko, obb, nullptr, out);
}
